// Upsample_35699768165019
// MI455X (gfx1250) — matrix operand order repaired and checked by reading, not compiled, not run on hardware
//
#include <hip/hip_runtime.h>
#include <hip/hip_bf16.h>
#include <stddef.h>

typedef __attribute__((ext_vector_type(16))) __bf16 v16bf;
typedef __attribute__((ext_vector_type(8)))  float  v8f;

#define NB   2
#define CI   128
#define CO   128
#define DIN  16
#define HIN  32
#define WIN  32
#define DOUT 32
#define HOUT 64
#define WOUT 64

// ---------------------------------------------------------------------------
// Kernel 1: fold the 3x3x3 kernel into 8 parity-class effective weights,
// emitted directly in WMMA bf16 A-fragment order:
//   wfrag[parity(8)][co_tile(8)][k_step(32)][lane(32)][elem(16)]  (bf16)
// K index = ci*8 + (jd*4 + jh*2 + jw); K within a 32-step per ISA A layout:
//   base = (lane<16)?0:8 ; e<8 -> K=base+e ; e>=8 -> K=base+e+8
// ---------------------------------------------------------------------------
__global__ __launch_bounds__(256) void weff_kernel(const float* __restrict__ Wk,
                                                   __bf16* __restrict__ wfrag) {
    int idx = blockIdx.x * 256 + threadIdx.x;   // 0 .. 1048575
    int e    = idx & 15;
    int lane = (idx >> 4)  & 31;
    int kk   = (idx >> 9)  & 31;
    int t    = (idx >> 14) & 7;
    int par  = (idx >> 17) & 7;
    int pd = (par >> 2) & 1, ph = (par >> 1) & 1, pw = par & 1;

    int co   = t * 16 + (lane & 15);
    int base = (lane < 16) ? 0 : 8;
    int kin  = base + e + ((e >= 8) ? 8 : 0);
    int K    = kk * 32 + kin;
    int ci   = K >> 3;
    int j    = K & 7;
    int jd = (j >> 2) & 1, jh = (j >> 1) & 1, jw = j & 1;

    // tap ranges per (parity, input-offset-index):
    // (0,0)->{0} (0,1)->{1,2} (1,0)->{0,1} (1,1)->{2}
    const int lo[2][2] = {{0, 1}, {0, 2}};
    const int hi[2][2] = {{0, 2}, {1, 2}};

    float s = 0.0f;
    for (int td = lo[pd][jd]; td <= hi[pd][jd]; ++td)
        for (int th = lo[ph][jh]; th <= hi[ph][jh]; ++th)
            for (int tw = lo[pw][jw]; tw <= hi[pw][jw]; ++tw)
                s += Wk[(((co * CI + ci) * 3 + td) * 3 + th) * 3 + tw];

    wfrag[idx] = (__bf16)s;
}

// ---------------------------------------------------------------------------
// Kernel 2: one workgroup per (b, di, hi, w-tile of 16). 8 waves, one per
// output parity class (pd,ph,pw). Each wave computes a [co=128] x [n=16]
// output tile as a K=1024 GEMM: 256 v_wmma_f32_16x16x32_bf16 per wave.
// All loads use immediate offsets from rolling pointers; A fragments are
// pipelined 4 deep, B fragments double-buffered; last K-step peeled.
// ---------------------------------------------------------------------------
__global__ __launch_bounds__(256) void upconv_kernel(const float*  __restrict__ x,
                                                     const __bf16* __restrict__ wfrag,
                                                     const float*  __restrict__ bias,
                                                     float*        __restrict__ out) {
    // LDS input patch: [ci=128][d=3][h=3][w=18] bf16 (halo of 1, zero padded)
    __shared__ __bf16 Xlds[CI * 3 * 3 * 18];   // 41472 bytes
    __shared__ float  Blds[CO];

    int blk = blockIdx.x;
    int wt  = blk & 1;
    int hi  = (blk >> 1) & 31;
    int di  = (blk >> 6) & 15;
    int b   = (blk >> 10);
    int tid = threadIdx.x;
    int wib = wt * 16;

    // -------- stage x (f32 -> bf16) into LDS with boundary zeros --------
    for (int e = tid; e < CI * 3 * 3 * 18; e += 256) {
        int ww = e % 18;
        int r  = e / 18;
        int hh = r % 3;
        int dd = (r / 3) % 3;
        int ci = r / 9;
        int dg = di - 1 + dd;
        int hg = hi - 1 + hh;
        int wg = wib - 1 + ww;
        float v = 0.0f;
        if ((unsigned)dg < (unsigned)DIN && (unsigned)hg < (unsigned)HIN &&
            (unsigned)wg < (unsigned)WIN)
            v = x[(((size_t)(b * CI + ci) * DIN + dg) * HIN + hg) * WIN + wg];
        Xlds[e] = (__bf16)v;
    }
    if (tid < CO) Blds[tid] = bias[tid];
    __syncthreads();

    int wave = tid >> 5;                 // parity class 0..7
    int lane = tid & 31;
    int pd = (wave >> 2) & 1, ph = (wave >> 1) & 1, pw = wave & 1;
    int n     = lane & 15;               // output column (w position in tile)
    int khalf = (lane < 16) ? 0 : 16;    // B-fragment K half per lane group

    // Rolling per-lane pointers: all in-loop loads use immediate offsets.
    // A(t,kk) = wp + t*16384 + kk*512 (elements); wp advances 512/K-step.
    const __bf16* wp = wfrag + ((size_t)wave * 8 * 32 * 32 + lane) * 16;
    // B gather base: lane/parity terms folded in; e-terms are constants;
    // kk term is a +648-element bump per K-step.
    const __bf16* xb = Xlds + (pd * 54 + ph * 18 + pw + n) + (khalf >> 4) * 162;

    auto buildB = [&]() -> v16bf {
        v16bf bv;
#pragma unroll
        for (int e = 0; e < 16; ++e) {
            int j   = e & 7;
            int off = (e >> 3) * 324 + ((j >> 2) & 1) * 54 +
                      ((j >> 1) & 1) * 18 + (j & 1);     // compile-time const
            bv[e] = xb[off];
        }
        return bv;
    };

    v8f acc[8];
#pragma unroll
    for (int t = 0; t < 8; ++t) acc[t] = (v8f){};

    // Pipeline prologue: 4 A fragments in flight + first B fragment.
    v16bf a0 = *(const v16bf*)(wp);
    v16bf a1 = *(const v16bf*)(wp + 16384);
    v16bf a2 = *(const v16bf*)(wp + 2 * 16384);
    v16bf a3 = *(const v16bf*)(wp + 3 * 16384);
    v16bf bcur = buildB();
    xb += 648;

    for (int kk = 0; kk < 31; ++kk) {
        v16bf bnext = buildB();            // gather B(kk+1) behind the WMMAs
#pragma unroll
        for (int t = 0; t < 8; ++t) {
            v16bf a = a0;                  // depth-4 rotation
            a0 = a1; a1 = a2; a2 = a3;
            int s = t + 4;                 // prefetch 4 steps ahead
            a3 = *(const v16bf*)(wp + (s & 7) * 16384 + (s >> 3) * 512);
            acc[t] = __builtin_amdgcn_wmma_f32_16x16x32_bf16(
                false, a, false, bcur, (short)0, acc[t], false, false);
        }
        bcur = bnext;
        wp += 512;
        xb += 648;
    }
    // Peeled last K-step (kk = 31): only in-bounds prefetches (t < 4).
#pragma unroll
    for (int t = 0; t < 8; ++t) {
        v16bf a = a0;
        a0 = a1; a1 = a2; a2 = a3;
        if (t < 4)
            a3 = *(const v16bf*)(wp + (t + 4) * 16384);
        acc[t] = __builtin_amdgcn_wmma_f32_16x16x32_bf16(
            false, a, false, bcur, (short)0, acc[t], false, false);
    }

    // -------- store: D rows = co, columns = spatial n --------
    int mrow = (lane < 16) ? 0 : 8;
    int od = 2 * di + pd;
    int oh = 2 * hi + ph;
#pragma unroll
    for (int t = 0; t < 8; ++t) {
#pragma unroll
        for (int g = 0; g < 8; ++g) {
            int co = t * 16 + mrow + g;
            float v = acc[t][g] + Blds[co];
            size_t oidx = ((((size_t)b * CO + co) * DOUT + od) * HOUT + oh) * WOUT +
                          (size_t)(2 * (wib + n) + pw);
            out[oidx] = v;
        }
    }
}

// ---------------------------------------------------------------------------
extern "C" void kernel_launch(void* const* d_in, const int* in_sizes, int n_in,
                              void* d_out, int out_size, void* d_ws, size_t ws_size,
                              hipStream_t stream) {
    const float* x    = (const float*)d_in[0];
    const float* Wk   = (const float*)d_in[1];
    const float* bias = (const float*)d_in[2];
    float*       out  = (float*)d_out;
    __bf16*      wfrag = (__bf16*)d_ws;   // 8*128*1024 bf16 = 2 MB

    // 8 parities * 8 co_tiles * 32 ksteps * 32 lanes * 16 elems = 1048576
    weff_kernel<<<4096, 256, 0, stream>>>(Wk, wfrag);

    // 2 * 16 * 32 * 2 = 2048 blocks, 256 threads (8 waves = 8 parity classes)
    upconv_kernel<<<2048, 256, 0, stream>>>(x, wfrag, bias, out);
}
